// GAT_30150670418388
// MI455X (gfx1250) — compile-verified, matrix operand order checked by reading, not run on hardware
//
#include <hip/hip_runtime.h>

typedef __attribute__((ext_vector_type(16))) __bf16 v16bf;
typedef __attribute__((ext_vector_type(8)))  float  v8f;

// ---------------------------------------------------------------- utilities

__global__ __launch_bounds__(256) void fill_f32_k(float* p, float v, long long n) {
  long long i = (long long)blockIdx.x * 256 + threadIdx.x;
  if (i < n) p[i] = v;
}

__device__ __forceinline__ void atomic_max_f32(float* addr, float v) {
  // classic signed/unsigned punning trick; init value must be -inf
  if (v >= 0.0f) atomicMax((int*)addr, __float_as_int(v));
  else           atomicMin((unsigned int*)addr, __float_as_uint(v));
}

// ------------------------------------------------- WMMA fragment pre-packing
// A-fragment order (bf16 16x32 tile): frag[(tm*Kt+ks)*32 + lane][i]
//   m = tm*16 + (lane&15), k = ks*32 + 8*(lane>>4) + (i<8 ? i : i+8)
__global__ __launch_bounds__(256) void pack_a_k(const float* X, __bf16* Ap, int M, int K) {
  long long o = (long long)blockIdx.x * 256 + threadIdx.x;
  long long total = (long long)M * K;
  if (o >= total) return;
  int i    = (int)(o & 15);
  int lane = (int)((o >> 4) & 31);
  long long t = o >> 9;                 // = tm*Kt + ks
  int Kt = K >> 5;
  int ks = (int)(t % Kt);
  int tm = (int)(t / Kt);
  int m = tm * 16 + (lane & 15);
  int k = ks * 32 + ((lane >> 4) << 3) + (i < 8 ? i : i + 8);
  Ap[o] = (__bf16)X[(long long)m * K + k];
}

// B-fragment order from row-major W[K x N]; layout [tn][Kt][32 lanes][16]
__global__ __launch_bounds__(256) void pack_b_k(const float* W, __bf16* Bp, int K, int N) {
  long long o = (long long)blockIdx.x * 256 + threadIdx.x;
  long long total = (long long)K * N;
  if (o >= total) return;
  int i    = (int)(o & 15);
  int lane = (int)((o >> 4) & 31);
  long long t = o >> 9;                 // = tn*Kt + ks
  int Kt = K >> 5;
  int ks = (int)(t % Kt);
  int tn = (int)(t / Kt);
  int n = tn * 16 + (lane & 15);
  int k = ks * 32 + ((lane >> 4) << 3) + (i < 8 ? i : i + 8);
  Bp[o] = (__bf16)W[(long long)k * N + n];
}

// ------------------------------------------------------------- WMMA GEMM
// C[M x N] = A[M x K] * B[K x N], N = NT*16. One wave owns a 16-row strip and
// all NT column tiles: each A fragment is loaded once and feeds NT WMMAs.
template <int NT>
__global__ __launch_bounds__(256) void wmma_gemm_strip_k(const __bf16* __restrict__ Ap,
                                                         const __bf16* __restrict__ Bp,
                                                         float* __restrict__ C,
                                                         int Mt, int Kt) {
  const int N = NT * 16;
  int wave = threadIdx.x >> 5;
  int lane = threadIdx.x & 31;
  long long tm = (long long)blockIdx.x * 8 + wave;
  if (tm >= Mt) return;                  // whole-wave exit: EXEC all-1 for WMMA
  const v16bf* Af = (const v16bf*)Ap;
  const v16bf* Bf = (const v16bf*)Bp;
  v8f zero = {};
  v8f acc[NT];
  #pragma unroll
  for (int t = 0; t < NT; ++t) acc[t] = zero;
  for (int ks = 0; ks < Kt; ++ks) {
    v16bf a = Af[(tm * Kt + ks) * 32 + lane];
    #pragma unroll
    for (int t = 0; t < NT; ++t) {
      v16bf b = Bf[((long long)t * Kt + ks) * 32 + lane];
      acc[t] = __builtin_amdgcn_wmma_f32_16x16x32_bf16(
          false, a, false, b, (short)0, acc[t], false, false);
    }
  }
  int colb = lane & 15;
  long long rbase = tm * 16 + ((lane >> 4) << 3);
  #pragma unroll
  for (int t = 0; t < NT; ++t) {
    #pragma unroll
    for (int r = 0; r < 8; ++r)
      C[(rbase + r) * N + t * 16 + colb] = acc[t][r];
  }
}

// --------------------------------------------------------------- edge phase
// wave per edge; FPL = features per lane (C = 32*FPL). FPL=4 -> one b128 row
// gather per matrix per lane; coalesced 512B per wave from L2-resident tables.
template <int FPL>
__global__ __launch_bounds__(256) void edge_logits_k(const float* __restrict__ xl,
                                                     const float* __restrict__ xr,
                                                     const float* __restrict__ att,
                                                     const int* __restrict__ esrc,
                                                     const int* __restrict__ edst,
                                                     int E, int Nn,
                                                     float* __restrict__ logits,
                                                     float* __restrict__ mmax) {
  const int C = FPL * 32;
  int wave = threadIdx.x >> 5, lane = threadIdx.x & 31;
  long long e = (long long)blockIdx.x * 8 + wave;
  long long EA = (long long)E + Nn;
  if (e >= EA) return;
  int s, d;
  if (e < E) { s = esrc[e]; d = edst[e]; } else { s = d = (int)(e - E); }
  float part = 0.0f;
  if constexpr (FPL == 4) {
    float4 l4 = ((const float4*)(xl + (long long)s * C))[lane];
    float4 r4 = ((const float4*)(xr + (long long)d * C))[lane];
    float4 a4 = ((const float4*)att)[lane];
    float v;
    v = l4.x + r4.x; v = fmaxf(v, 0.2f * v); part += v * a4.x;
    v = l4.y + r4.y; v = fmaxf(v, 0.2f * v); part += v * a4.y;
    v = l4.z + r4.z; v = fmaxf(v, 0.2f * v); part += v * a4.z;
    v = l4.w + r4.w; v = fmaxf(v, 0.2f * v); part += v * a4.w;
  } else {
    float v = xl[(long long)s * C + lane] + xr[(long long)d * C + lane];
    v = fmaxf(v, 0.2f * v);               // leaky_relu, slope 0.2
    part = v * att[lane];
  }
  #pragma unroll
  for (int m = 16; m >= 1; m >>= 1) part += __shfl_xor(part, m, 32);
  if (lane == 0) {
    logits[e] = part;
    atomic_max_f32(&mmax[d], part);
  }
}

template <int FPL>
__global__ __launch_bounds__(256) void edge_agg_k(const float* __restrict__ xl,
                                                  const float* __restrict__ logits,
                                                  const float* __restrict__ mmax,
                                                  const int* __restrict__ esrc,
                                                  const int* __restrict__ edst,
                                                  int E, int Nn,
                                                  float* __restrict__ denom,
                                                  float* __restrict__ acc) {
  const int C = FPL * 32;
  int wave = threadIdx.x >> 5, lane = threadIdx.x & 31;
  long long e = (long long)blockIdx.x * 8 + wave;
  long long EA = (long long)E + Nn;
  if (e >= EA) return;
  int s, d;
  if (e < E) { s = esrc[e]; d = edst[e]; } else { s = d = (int)(e - E); }
  float ev = __expf(logits[e] - mmax[d]);
  if (lane == 0) atomicAdd(&denom[d], ev);
  if constexpr (FPL == 4) {
    float4 l4 = ((const float4*)(xl + (long long)s * C))[lane];
    float* ap = acc + (long long)d * C + lane * 4;
    atomicAdd(ap + 0, l4.x * ev);
    atomicAdd(ap + 1, l4.y * ev);
    atomicAdd(ap + 2, l4.z * ev);
    atomicAdd(ap + 3, l4.w * ev);
  } else {
    atomicAdd(acc + (long long)d * C + lane, xl[(long long)s * C + lane] * ev);
  }
}

// finalize: divide by denom, add bias, ReLU (both layers end in ReLU)
__global__ __launch_bounds__(256) void finalize_k(const float* __restrict__ acc,
                                                  const float* __restrict__ denom,
                                                  const float* __restrict__ bias,
                                                  float* __restrict__ out,
                                                  long long total, int C) {
  long long i = (long long)blockIdx.x * 256 + threadIdx.x;
  if (i >= total) return;
  long long n = i / C;
  int c = (int)(i % C);
  out[i] = fmaxf(acc[i] / (denom[n] + 1e-16f) + bias[c], 0.0f);
}

// ------------------------------------------------------------------- driver

extern "C" void kernel_launch(void* const* d_in, const int* in_sizes, int n_in,
                              void* d_out, int out_size, void* d_ws, size_t ws_size,
                              hipStream_t stream) {
  const int F_IN = 128, H = 128, OUT = 32;
  const float* x    = (const float*)d_in[0];
  const int*   eidx = (const int*)  d_in[1];
  const float* Wl1  = (const float*)d_in[2];
  const float* Wr1  = (const float*)d_in[3];
  const float* att1 = (const float*)d_in[4];
  const float* b1   = (const float*)d_in[5];
  const float* Wl2  = (const float*)d_in[6];
  const float* Wr2  = (const float*)d_in[7];
  const float* att2 = (const float*)d_in[8];
  const float* b2   = (const float*)d_in[9];
  float* out = (float*)d_out;

  const int Nn = in_sizes[0] / F_IN;        // 50000
  const int E  = in_sizes[1] / 2;           // 1600000
  const long long EA = (long long)E + Nn;   // with self loops
  const int* esrc = eidx;
  const int* edst = eidx + E;

  // workspace carve-out (256B aligned)
  char* ws = (char*)d_ws;
  size_t off = 0;
  auto carve = [&](size_t bytes) -> char* {
    char* p = ws + off;
    off += (bytes + 255) & ~(size_t)255;
    return p;
  };
  __bf16* apack = (__bf16*)carve((size_t)Nn * F_IN * sizeof(__bf16)); // reused by layer 2
  __bf16* bpl1  = (__bf16*)carve((size_t)F_IN * H   * sizeof(__bf16));
  __bf16* bpr1  = (__bf16*)carve((size_t)F_IN * H   * sizeof(__bf16));
  __bf16* bpl2  = (__bf16*)carve((size_t)H * OUT    * sizeof(__bf16));
  __bf16* bpr2  = (__bf16*)carve((size_t)H * OUT    * sizeof(__bf16));
  float*  xl    = (float*) carve((size_t)Nn * H * sizeof(float));     // layer2 uses Nn*OUT prefix
  float*  xr    = (float*) carve((size_t)Nn * H * sizeof(float));
  float*  logit = (float*) carve((size_t)EA * sizeof(float));
  float*  mmax  = (float*) carve((size_t)Nn * sizeof(float));
  float*  denom = (float*) carve((size_t)Nn * sizeof(float));
  float*  accb  = (float*) carve((size_t)Nn * H * sizeof(float));
  float*  hbuf  = (float*) carve((size_t)Nn * H * sizeof(float));

  auto g256 = [](long long n) { return (unsigned)((n + 255) / 256); };
  auto gwav = [](long long n) { return (unsigned)((n + 7) / 8); };

  // ---- pack weights (tiny)
  pack_b_k<<<g256((long long)F_IN * H), 256, 0, stream>>>(Wl1, bpl1, F_IN, H);
  pack_b_k<<<g256((long long)F_IN * H), 256, 0, stream>>>(Wr1, bpr1, F_IN, H);
  pack_b_k<<<g256((long long)H * OUT),  256, 0, stream>>>(Wl2, bpl2, H, OUT);
  pack_b_k<<<g256((long long)H * OUT),  256, 0, stream>>>(Wr2, bpr2, H, OUT);

  const int Mt = Nn / 16;

  // ================= layer 1 (C = 128) =================
  {
    const int C = H, K = F_IN, Kt = K / 32;
    pack_a_k<<<g256((long long)Nn * K), 256, 0, stream>>>(x, apack, Nn, K);
    wmma_gemm_strip_k<8><<<gwav(Mt), 256, 0, stream>>>(apack, bpl1, xl, Mt, Kt);
    wmma_gemm_strip_k<8><<<gwav(Mt), 256, 0, stream>>>(apack, bpr1, xr, Mt, Kt);

    fill_f32_k<<<g256(Nn), 256, 0, stream>>>(mmax, -INFINITY, Nn);
    fill_f32_k<<<g256(Nn), 256, 0, stream>>>(denom, 0.0f, Nn);
    fill_f32_k<<<g256((long long)Nn * C), 256, 0, stream>>>(accb, 0.0f, (long long)Nn * C);

    edge_logits_k<4><<<gwav(EA), 256, 0, stream>>>(xl, xr, att1, esrc, edst, E, Nn, logit, mmax);
    edge_agg_k<4><<<gwav(EA), 256, 0, stream>>>(xl, logit, mmax, esrc, edst, E, Nn, denom, accb);
    finalize_k<<<g256((long long)Nn * C), 256, 0, stream>>>(accb, denom, b1, hbuf,
                                                            (long long)Nn * C, C);
  }

  // ================= layer 2 (C = 32) =================
  {
    const int C = OUT, K = H, Kt = K / 32;
    pack_a_k<<<g256((long long)Nn * K), 256, 0, stream>>>(hbuf, apack, Nn, K);
    wmma_gemm_strip_k<2><<<gwav(Mt), 256, 0, stream>>>(apack, bpl2, xl, Mt, Kt);
    wmma_gemm_strip_k<2><<<gwav(Mt), 256, 0, stream>>>(apack, bpr2, xr, Mt, Kt);

    fill_f32_k<<<g256(Nn), 256, 0, stream>>>(mmax, -INFINITY, Nn);
    fill_f32_k<<<g256(Nn), 256, 0, stream>>>(denom, 0.0f, Nn);
    fill_f32_k<<<g256((long long)Nn * C), 256, 0, stream>>>(accb, 0.0f, (long long)Nn * C);

    edge_logits_k<1><<<gwav(EA), 256, 0, stream>>>(xl, xr, att2, esrc, edst, E, Nn, logit, mmax);
    edge_agg_k<1><<<gwav(EA), 256, 0, stream>>>(xl, logit, mmax, esrc, edst, E, Nn, denom, accb);
    finalize_k<<<g256((long long)Nn * C), 256, 0, stream>>>(accb, denom, b2, out,
                                                            (long long)Nn * C, C);
  }
}
